// CausalAttentionOpCUDA_79972291052162
// MI455X (gfx1250) — matrix operand order repaired and checked by reading, not compiled, not run on hardware
//
#include <hip/hip_runtime.h>
#include <hip/hip_bf16.h>
#include <math.h>

// Problem constants from the reference
#define D      256
#define RANK   64
#define NBATCH 4096
#define GSTRIDE 257   // padded LDS row stride: (i*257+j)&63 = (i+j)&63 -> conflict-free

typedef __attribute__((ext_vector_type(16))) __bf16 v16bf;
typedef __attribute__((ext_vector_type(8)))  float  v8f;

// Batch-independent precomputed factors (recomputed every call; deterministic)
__device__ float g_G[D * D];  // 0.125 * Q_emb @ K_emb^T   (256 KB, L2/LDS resident)
__device__ float g_w[D];      // V_emb @ out_proj

// ---------------------------------------------------------------------------
// Kernel A: G = scale * Q_emb(256x64) @ K_emb(256x64)^T  via WMMA bf16.
// One wave per 16x16 output tile; K-dim = 64 -> two v_wmma_f32_16x16x32_bf16.
// ---------------------------------------------------------------------------
__global__ __launch_bounds__(32)
void build_G_kernel(const float* __restrict__ Q, const float* __restrict__ K) {
  const int lane = threadIdx.x & 31;
  const int ti   = (blockIdx.x >> 4) << 4;   // output row-tile base
  const int tj   = (blockIdx.x & 15) << 4;   // output col-tile base
  const int m    = lane & 15;                // row (A) / col (B) within tile
  const int hs   = lane >> 4;                // lane-half selector

  v8f c = {};
  #pragma unroll
  for (int kc = 0; kc < RANK; kc += 32) {
    v16bf a, b;
    // A 16x32 (MxK) bf16 wave32 layout:
    //   lane<16 : M=lane,    elems 0..7 -> K=kc+0..7,   elems 8..15 -> K=kc+16..23
    //   lane>=16: M=lane-16, elems 0..7 -> K=kc+8..15,  elems 8..15 -> K=kc+24..31
    const float* qrow = Q + (ti + m) * RANK + kc;
    #pragma unroll
    for (int e = 0; e < 8; ++e) {
      a[e]     = (__bf16)qrow[hs * 8 + e];
      a[e + 8] = (__bf16)qrow[16 + hs * 8 + e];
    }
    // B 32x16 (KxN) bf16 wave32 layout: N = lane&15; same K order as A (elems 0..7 -> K=8*half+0..7, 8..15 -> K=16+8*half+0..7).
    // B[k][n] = K_emb[tj+n][kc+k]  (so C = Q_emb K_emb^T)
    const float* krow = K + (tj + m) * RANK + kc;   // n == m here
    #pragma unroll
    for (int e = 0; e < 16; ++e)
      b[e] = (__bf16)krow[(e & 8) * 2 + hs * 8 + (e & 7)];

    c = __builtin_amdgcn_wmma_f32_16x16x32_bf16(false, a, false, b,
                                                (short)0, c, false, false);
  }

  const float scale = 0.125f;  // 1/sqrt(RANK)
  #pragma unroll
  for (int v = 0; v < 8; ++v) {
    const int row = ti + hs * 8 + v;   // C/D layout: elem v -> M = v + 8*half
    const int col = tj + m;
    g_G[row * D + col] = c[v] * scale;
  }
}

// ---------------------------------------------------------------------------
// Kernel B: w[j] = dot(V_emb[j,:], out_proj)   (tiny)
// ---------------------------------------------------------------------------
__global__ __launch_bounds__(256)
void build_w_kernel(const float* __restrict__ V, const float* __restrict__ op) {
  const int j = threadIdx.x;
  float acc = 0.f;
  #pragma unroll 8
  for (int r = 0; r < RANK; ++r)
    acc = fmaf(V[j * RANK + r], op[r], acc);
  g_w[j] = acc;
}

// ---------------------------------------------------------------------------
// Kernel C: per-batch causal softmax using the collapsed form.
//   s[i,j] = x_i * x_j * G[i,j]  (j<=i),  out[i] = x_i + gate * softmax(s)·(x⊙w)
// G staged once per workgroup into 320KB LDS (padded stride); grid-stride over
// batches amortizes the stage. Two passes over the causal prefix: pass1 max
// (no exp), pass2 single exp per element + fused weighted accumulate.
// ---------------------------------------------------------------------------
__global__ __launch_bounds__(256)
void attn_kernel(const float* __restrict__ x, const float* __restrict__ gate_logit,
                 float* __restrict__ out) {
  extern __shared__ float smem[];
  float* sG = smem;                  // GSTRIDE * D floats
  float* sx = smem + GSTRIDE * D;    // D floats
  float* su = sx + D;                // D floats

  const int t = threadIdx.x;

  // Stage G into LDS with padded rows (coalesced global reads, conflict-free LDS)
  for (int idx = t; idx < D * D; idx += 256) {
    const int i = idx >> 8, j = idx & (D - 1);
    sG[i * GSTRIDE + j] = g_G[idx];
  }

  const float gate = 1.f / (1.f + __expf(-gate_logit[0]));
  const float wj_t = g_w[t];

  for (int b = blockIdx.x; b < NBATCH; b += gridDim.x) {
    __syncthreads();                       // staging done / previous batch done
    const float xv = x[b * D + t];
    sx[t] = xv;
    su[t] = xv * wj_t;
    __syncthreads();

    const int i = t;                       // one row per thread
    const float xi = sx[i];
    const float* __restrict__ grow = sG + i * GSTRIDE;

    // pass 1: causal max (no exp)
    float mx = -__builtin_inff();
    for (int j = 0; j <= i; ++j) {
      const float v = xi * sx[j] * grow[j];
      mx = fmaxf(mx, v);
    }
    // pass 2: one exp per element, fused sum + weighted accumulate
    float s = 0.f, o = 0.f;
    for (int j = 0; j <= i; ++j) {
      const float v = xi * sx[j] * grow[j];
      const float e = __expf(v - mx);
      s += e;
      o = fmaf(e, su[j], o);
    }
    out[b * D + t] = xi + gate * (o / s);
  }
}

// ---------------------------------------------------------------------------
extern "C" void kernel_launch(void* const* d_in, const int* in_sizes, int n_in,
                              void* d_out, int out_size, void* d_ws, size_t ws_size,
                              hipStream_t stream) {
  (void)in_sizes; (void)n_in; (void)out_size; (void)d_ws; (void)ws_size;
  const float* x  = (const float*)d_in[0];
  const float* Q  = (const float*)d_in[1];
  const float* K  = (const float*)d_in[2];
  const float* V  = (const float*)d_in[3];
  const float* op = (const float*)d_in[4];
  const float* gl = (const float*)d_in[5];
  float* out = (float*)d_out;

  build_G_kernel<<<(D / 16) * (D / 16), 32, 0, stream>>>(Q, K);
  build_w_kernel<<<1, 256, 0, stream>>>(V, op);

  const size_t shmem = (size_t)(GSTRIDE * D + 2 * D) * sizeof(float);  // ~265 KB < 320 KB
  hipFuncSetAttribute((const void*)attn_kernel,
                      hipFuncAttributeMaxDynamicSharedMemorySize, (int)shmem);
  attn_kernel<<<512, 256, shmem, stream>>>(x, gl, out);
}
